// InvariantPointAttention_52467320488362
// MI455X (gfx1250) — matrix operand order repaired and checked by reading, not compiled, not run on hardware
//
#include <hip/hip_runtime.h>
#include <math.h>

// ---------------- problem constants ----------------
#define NRES   512
#define NODE_D 256
#define EDGE_D 128
#define HEADS  8

static constexpr float SCALAR_SCALE = 0.14433756729740643f;  // (3*16)^-0.5
static constexpr float POINT_SCALE  = 0.13608276348795434f;  // (3*4*4.5)^-0.5
static constexpr float PAIR_SCALE   = 0.57735026918962580f;  // 3^-0.5
static constexpr float EPSV         = 1e-8f;

typedef _Float16 v16h __attribute__((ext_vector_type(16)));
typedef float    v8f  __attribute__((ext_vector_type(8)));
typedef float    f4   __attribute__((ext_vector_type(4), aligned(16)));

// ---------------- WMMA wrapper (v_wmma_f32_16x16x32_f16) --------------------
__device__ __forceinline__ v8f wmma_f16(v16h a, v16h b, v8f c) {
  return __builtin_amdgcn_wmma_f32_16x16x32_f16(false, a, false, b, (short)0, c,
                                                false, false);
}

// A fragment, FULL tile: logical A[m][k] row-major, rows m0..m0+15,
// k in [k0,k0+32). lane&15 = M row; lanes<16 hold K {0..15},
// lanes>=16 hold K {16..31}. 4x b128 loads per lane.
__device__ __forceinline__ v16h frag_a_full(const float* __restrict__ A, const int lda,
                                            int m0, int k0) {
  const int lane = threadIdx.x & 31;
  const float* p = A + (size_t)(m0 + (lane & 15)) * (size_t)lda + k0 + ((lane >> 4) * 16);
  const f4 x0 = *(const f4*)(p);
  const f4 x1 = *(const f4*)(p + 4);
  const f4 y0 = *(const f4*)(p + 8);
  const f4 y1 = *(const f4*)(p + 12);
  v16h f;
#pragma unroll
  for (int i = 0; i < 4; ++i) {
    f[i]      = (_Float16)x0[i];
    f[i + 4]  = (_Float16)x1[i];
    f[i + 8]  = (_Float16)y0[i];
    f[i + 12] = (_Float16)y1[i];
  }
  return f;
}

// A fragment with row index clamped to [0,8) (dup rows feed discarded D rows).
__device__ __forceinline__ v16h frag_a_clamp8(const float* __restrict__ A, const int lda,
                                              int k0) {
  const int lane = threadIdx.x & 31;
  int m = lane & 15; if (m >= 8) m -= 8;
  const float* p = A + (size_t)m * (size_t)lda + k0 + ((lane >> 4) * 16);
  const f4 x0 = *(const f4*)(p);
  const f4 x1 = *(const f4*)(p + 4);
  const f4 y0 = *(const f4*)(p + 8);
  const f4 y1 = *(const f4*)(p + 12);
  v16h f;
#pragma unroll
  for (int i = 0; i < 4; ++i) {
    f[i]      = (_Float16)x0[i];
    f[i + 4]  = (_Float16)x1[i];
    f[i + 8]  = (_Float16)y0[i];
    f[i + 12] = (_Float16)y1[i];
  }
  return f;
}

// B fragment from row-major KxN storage (strided; constant-offset dword loads).
__device__ __forceinline__ v16h frag_b_full(const float* __restrict__ B, const int ldb,
                                            int k0, int n0) {
  const int lane = threadIdx.x & 31;
  const float* p = B + (size_t)(k0 + ((lane >> 4) * 16)) * (size_t)ldb + n0 + (lane & 15);
  v16h f;
#pragma unroll
  for (int i = 0; i < 16; ++i) f[i] = (_Float16)p[i * ldb];  // int offsets -> immediates
  return f;
}

// B fragment from transposed storage: B[k][n] = Bt[n][k]; contiguous in k,
// 4x b128 loads per lane.
__device__ __forceinline__ v16h frag_bT_full(const float* __restrict__ Bt, const int ldbt,
                                             int k0, int n0) {
  const int lane = threadIdx.x & 31;
  const float* p = Bt + (size_t)(n0 + (lane & 15)) * (size_t)ldbt + k0 + ((lane >> 4) * 16);
  const f4 x0 = *(const f4*)(p);
  const f4 x1 = *(const f4*)(p + 4);
  const f4 x2 = *(const f4*)(p + 8);
  const f4 x3 = *(const f4*)(p + 12);
  v16h f;
#pragma unroll
  for (int i = 0; i < 4; ++i) {
    f[i]      = (_Float16)x0[i];
    f[i + 4]  = (_Float16)x1[i];
    f[i + 8]  = (_Float16)x2[i];
    f[i + 12] = (_Float16)x3[i];
  }
  return f;
}

__device__ __forceinline__ float softplusf(float x) {
  return (x > 20.f) ? x : log1pf(__expf(x));
}

// ---------------- K0: init -------------------------------------------------
// zr  : [qs|ks|qp|kp|vpT] = 589824 floats zeroed (padded K dims / padded rows)
// WT  : fused weight transpose [672][256] for K1
// WpbT: [16][128] (rows 8..15 zero) for K3
// WoutT: [256][1280] for K9
__global__ void k0_init(float* __restrict__ zr,
                        const float* __restrict__ Wsq, const float* __restrict__ Wsk,
                        const float* __restrict__ Wsv, const float* __restrict__ Wpq,
                        const float* __restrict__ Wpk, const float* __restrict__ Wpv,
                        float* __restrict__ WT,
                        const float* __restrict__ Wpb, float* __restrict__ WpbT,
                        const float* __restrict__ Wout, float* __restrict__ WoutT) {
  const int idx = blockIdx.x * blockDim.x + threadIdx.x;
  if (idx < 589824) zr[idx] = 0.f;
  if (idx < 672 * 256) {              // WT[c][k] = W*(k, c_local)
    const int c = idx >> 8, k = idx & 255;
    float v;
    if      (c < 128) v = Wsq[k * 128 + c];
    else if (c < 256) v = Wsk[k * 128 + (c - 128)];
    else if (c < 384) v = Wsv[k * 128 + (c - 256)];
    else if (c < 480) v = Wpq[k * 96 + (c - 384)];
    else if (c < 576) v = Wpk[k * 96 + (c - 480)];
    else              v = Wpv[k * 96 + (c - 576)];
    WT[idx] = v;
  }
  if (idx < 16 * 128) {               // WpbT[n][k]
    const int n = idx >> 7, k = idx & 127;
    WpbT[idx] = (n < 8) ? Wpb[k * 8 + n] : 0.f;
  }
  if (idx < 256 * 1280) {             // WoutT[n][k]
    const int n = idx / 1280, k = idx % 1280;
    WoutT[idx] = Wout[k * 256 + n];
  }
}

// ---------------- K1: fused projection GEMM ---------------------------------
// node(512x256) @ WT^T. qs/ks -> [H][N][32] padded, vsT -> [H][16][N],
// praw_* -> [N][96].
__global__ void k1_proj(const float* __restrict__ node, const float* __restrict__ WT,
                        float* __restrict__ qs, float* __restrict__ ks,
                        float* __restrict__ vsT, float* __restrict__ prq,
                        float* __restrict__ prk, float* __restrict__ prv) {
  const int gw = blockIdx.x * (blockDim.x >> 5) + (threadIdx.x >> 5);
  const int NT_COL = 42;  // 672/16
  if (gw >= (NRES / 16) * NT_COL) return;
  const int mt = gw / NT_COL, ct = gw % NT_COL;
  const int m0 = mt * 16, c0 = ct * 16;

  v8f acc = {};
#pragma unroll
  for (int k0 = 0; k0 < NODE_D; k0 += 32)
    acc = wmma_f16(frag_a_full(node, NODE_D, m0, k0),
                   frag_bT_full(WT, NODE_D, k0, c0), acc);

  int t, cbase;
  if      (c0 < 128) { t = 0; cbase = 0;   }
  else if (c0 < 256) { t = 1; cbase = 128; }
  else if (c0 < 384) { t = 2; cbase = 256; }
  else if (c0 < 480) { t = 3; cbase = 384; }
  else if (c0 < 576) { t = 4; cbase = 480; }
  else               { t = 5; cbase = 576; }
  const int lane = threadIdx.x & 31;
  const int n = lane & 15, rb = (lane >> 4) * 8;
  const int c = c0 - cbase + n;
  float* ds = (t == 0) ? qs : ks;
  float* dp = (t == 3) ? prq : (t == 4) ? prk : prv;
#pragma unroll
  for (int r = 0; r < 8; ++r) {
    const int m = m0 + rb + r;
    if (t < 2)       ds[(size_t)(c >> 4) * (NRES * 32) + (size_t)m * 32 + (c & 15)] = acc[r];
    else if (t == 2) vsT[(size_t)(c >> 4) * (16 * NRES) + (size_t)(c & 15) * NRES + m] = acc[r];
    else             dp[(size_t)m * 96 + c] = acc[r];
  }
}

// ---------------- K2: rotate/translate points, q2/k2 ------------------------
// qp/kp -> [H][N][32] (cols 12..31 zero), vpT -> [H][16][N] (rows 12..15 zero).
__global__ void k2_transform(const float* __restrict__ prq, const float* __restrict__ prk,
                             const float* __restrict__ prv, const float* __restrict__ rot,
                             const float* __restrict__ trans,
                             float* __restrict__ qp, float* __restrict__ kp,
                             float* __restrict__ vpT, float* __restrict__ q2,
                             float* __restrict__ k2) {
  const int idx = blockIdx.x * blockDim.x + threadIdx.x;
  if (idx >= NRES * HEADS) return;
  const int i = idx >> 3, h = idx & 7;
  float R[9], T[3];
#pragma unroll
  for (int e = 0; e < 9; ++e) R[e] = rot[(size_t)i * 9 + e];
#pragma unroll
  for (int e = 0; e < 3; ++e) T[e] = trans[(size_t)i * 3 + e];

  const float* src[3] = {prq, prk, prv};
  for (int s = 0; s < 3; ++s) {
    float sum2 = 0.f;
#pragma unroll
    for (int p = 0; p < 4; ++p) {
      const float l0 = src[s][(size_t)i * 96 + (h * 4 + p) * 3 + 0];
      const float l1 = src[s][(size_t)i * 96 + (h * 4 + p) * 3 + 1];
      const float l2 = src[s][(size_t)i * 96 + (h * 4 + p) * 3 + 2];
#pragma unroll
      for (int r = 0; r < 3; ++r) {
        const float w = R[r * 3 + 0] * l0 + R[r * 3 + 1] * l1 + R[r * 3 + 2] * l2 + T[r];
        if (s == 0)      qp[(size_t)h * (NRES * 32) + (size_t)i * 32 + p * 3 + r] = w;
        else if (s == 1) kp[(size_t)h * (NRES * 32) + (size_t)i * 32 + p * 3 + r] = w;
        else             vpT[(size_t)h * (16 * NRES) + (size_t)(p * 3 + r) * NRES + i] = w;
        sum2 += w * w;
      }
    }
    if (s == 0) q2[h * NRES + i] = sum2;
    if (s == 1) k2[h * NRES + i] = sum2;
  }
}

// ---------------- K3: pair bias (edge @ W_pb) -> logits ---------------------
// 4 M-tiles (64 edge rows) per wave; B fragments loaded once, reused 4x.
__global__ void k3_pairbias(const float* __restrict__ edge, const float* __restrict__ WpbT,
                            const float* __restrict__ bpb, float* __restrict__ L) {
  const int gw = blockIdx.x * (blockDim.x >> 5) + (threadIdx.x >> 5);
  if (gw >= (NRES * NRES) / 64) return;
  const int m0 = gw * 64;
  __builtin_prefetch(edge + (size_t)(m0 + 64) * EDGE_D, 0, 1);

  const v16h b0 = frag_bT_full(WpbT, EDGE_D, 0, 0);
  const v16h b1 = frag_bT_full(WpbT, EDGE_D, 32, 0);
  const v16h b2 = frag_bT_full(WpbT, EDGE_D, 64, 0);
  const v16h b3 = frag_bT_full(WpbT, EDGE_D, 96, 0);

  const int lane = threadIdx.x & 31;
  const int h = lane & 15, rb = (lane >> 4) * 8;
  const float bb = (h < HEADS) ? bpb[h] : 0.f;

#pragma unroll
  for (int mt = 0; mt < 4; ++mt) {
    const int m0t = m0 + mt * 16;
    v8f acc = {};
    acc = wmma_f16(frag_a_full(edge, EDGE_D, m0t, 0),  b0, acc);
    acc = wmma_f16(frag_a_full(edge, EDGE_D, m0t, 32), b1, acc);
    acc = wmma_f16(frag_a_full(edge, EDGE_D, m0t, 64), b2, acc);
    acc = wmma_f16(frag_a_full(edge, EDGE_D, m0t, 96), b3, acc);
    if (h < HEADS) {
#pragma unroll
      for (int r = 0; r < 8; ++r)
        L[(size_t)h * (NRES * NRES) + (m0t + rb + r)] = (acc[r] + bb) * PAIR_SCALE;
    }
  }
}

// ---------------- K4: scalar + point logits ---------------------------------
__global__ void k4_logits(const float* __restrict__ qs, const float* __restrict__ ks,
                          const float* __restrict__ qp, const float* __restrict__ kp,
                          const float* __restrict__ q2, const float* __restrict__ k2,
                          const float* __restrict__ point_weights,
                          float* __restrict__ L) {
  const int gw = blockIdx.x * (blockDim.x >> 5) + (threadIdx.x >> 5);
  if (gw >= HEADS * 32 * 32) return;
  const int h = gw >> 10, rem = gw & 1023;
  const int i0 = (rem >> 5) * 16, j0 = (rem & 31) * 16;

  v8f acc1 = {}, acc2 = {};
  acc1 = wmma_f16(frag_a_full(qs + (size_t)h * NRES * 32, 32, i0, 0),
                  frag_bT_full(ks + (size_t)h * NRES * 32, 32, 0, j0), acc1);
  acc2 = wmma_f16(frag_a_full(qp + (size_t)h * NRES * 32, 32, i0, 0),
                  frag_bT_full(kp + (size_t)h * NRES * 32, 32, 0, j0), acc2);

  const float pw = softplusf(point_weights[h]);
  const float pc = -0.5f * pw * POINT_SCALE;
  const int lane = threadIdx.x & 31;
  const int j = j0 + (lane & 15), rb = (lane >> 4) * 8;
  const float k2v = k2[h * NRES + j];
#pragma unroll
  for (int r = 0; r < 8; ++r) {
    const int i = i0 + rb + r;
    const float q2v = q2[h * NRES + i];
    float* p = L + (size_t)h * (NRES * NRES) + (size_t)i * NRES + j;
    *p += SCALAR_SCALE * acc1[r] + pc * (q2v + k2v - 2.f * acc2[r]);
  }
}

// ---------------- K5: row softmax (wave per (h,i) row) ----------------------
__global__ void k5_softmax(float* __restrict__ L) {
  const int row = blockIdx.x * (blockDim.x >> 5) + (threadIdx.x >> 5);
  if (row >= HEADS * NRES) return;
  const int lane = threadIdx.x & 31;
  float* p = L + (size_t)row * NRES;
  float v[16];
  float m = -1e30f;
#pragma unroll
  for (int t = 0; t < 16; ++t) { v[t] = p[lane + t * 32]; m = fmaxf(m, v[t]); }
#pragma unroll
  for (int off = 16; off > 0; off >>= 1) m = fmaxf(m, __shfl_xor(m, off, 32));
  float s = 0.f;
#pragma unroll
  for (int t = 0; t < 16; ++t) { v[t] = __expf(v[t] - m); s += v[t]; }
#pragma unroll
  for (int off = 16; off > 0; off >>= 1) s += __shfl_xor(s, off, 32);
  const float inv = 1.f / s;
#pragma unroll
  for (int t = 0; t < 16; ++t) p[lane + t * 32] = v[t] * inv;
}

// ---------------- K6: attn @ vs and attn @ vp (fused, A-frag reuse) ---------
__global__ void k6_apply_v(const float* __restrict__ L, const float* __restrict__ vsT,
                           const float* __restrict__ vpT, float* __restrict__ feat,
                           float* __restrict__ ptw) {
  const int gw = blockIdx.x * (blockDim.x >> 5) + (threadIdx.x >> 5);
  if (gw >= HEADS * 32) return;
  const int h = gw >> 5, i0 = (gw & 31) * 16;

  v8f acc_s = {}, acc_p = {};
#pragma unroll 4
  for (int k0 = 0; k0 < NRES; k0 += 32) {
    v16h a = frag_a_full(L + (size_t)h * NRES * NRES, NRES, i0, k0);
    acc_s = wmma_f16(a, frag_bT_full(vsT + (size_t)h * 16 * NRES, NRES, k0, 0), acc_s);
    acc_p = wmma_f16(a, frag_bT_full(vpT + (size_t)h * 16 * NRES, NRES, k0, 0), acc_p);
  }
  const int lane = threadIdx.x & 31;
  const int n = lane & 15, rb = (lane >> 4) * 8;
#pragma unroll
  for (int r = 0; r < 8; ++r) {
    const int i = i0 + rb + r;
    feat[(size_t)i * 1280 + h * 16 + n] = acc_s[r];  // res_scalar block [0,128)
    if (n < 12) ptw[(size_t)h * NRES * 12 + (size_t)i * 12 + n] = acc_p[r];
  }
}

// ---------------- K7: res_pair = attn-weighted edge gather ------------------
// Per residue i: (8x512) @ (512x128); second edge pass (L2-resident).
__global__ void k7_pair(const float* __restrict__ L, const float* __restrict__ edge,
                        float* __restrict__ feat) {
  const int gw = blockIdx.x * (blockDim.x >> 5) + (threadIdx.x >> 5);
  if (gw >= NRES * 8) return;
  const int i = gw >> 3, n0 = (gw & 7) * 16;
  const float* A = L + (size_t)i * NRES;              // rows = heads, lda = N*N
  const float* B = edge + (size_t)i * NRES * EDGE_D;  // (512 x 128)
  __builtin_prefetch(B + (size_t)n0, 0, 1);

  v8f acc = {};
#pragma unroll 4
  for (int k0 = 0; k0 < NRES; k0 += 32)
    acc = wmma_f16(frag_a_clamp8(A, NRES * NRES, k0),
                   frag_b_full(B, EDGE_D, k0, n0), acc);

  const int lane = threadIdx.x & 31;
  const int n = lane & 15, rb = (lane >> 4) * 8;
  if (rb == 0) {  // rows 8..15 duplicate rows 0..7 -> only lanes<16 store
#pragma unroll
    for (int r = 0; r < 8; ++r)
      feat[(size_t)i * 1280 + 256 + r * EDGE_D + n0 + n] = acc[r];
  }
}

// ---------------- K8: un-rotate points, norms -> feature buffer -------------
__global__ void k8_points_out(const float* __restrict__ ptw, const float* __restrict__ rot,
                              const float* __restrict__ trans, float* __restrict__ feat) {
  const int idx = blockIdx.x * blockDim.x + threadIdx.x;
  if (idx >= NRES * HEADS) return;
  const int i = idx >> 3, h = idx & 7;
  float R[9], T[3];
#pragma unroll
  for (int e = 0; e < 9; ++e) R[e] = rot[(size_t)i * 9 + e];
#pragma unroll
  for (int e = 0; e < 3; ++e) T[e] = trans[(size_t)i * 3 + e];
  float* fb = feat + (size_t)i * 1280;
#pragma unroll
  for (int p = 0; p < 4; ++p) {
    float w[3];
#pragma unroll
    for (int r = 0; r < 3; ++r)
      w[r] = ptw[(size_t)h * NRES * 12 + (size_t)i * 12 + p * 3 + r] - T[r];
    float s2 = 0.f;
#pragma unroll
    for (int c = 0; c < 3; ++c) {
      const float lc = w[0] * R[0 * 3 + c] + w[1] * R[1 * 3 + c] + w[2] * R[2 * 3 + c];
      fb[128 + h * 12 + p * 3 + c] = lc;       // res_pts block [128,224)
      s2 += lc * lc;
    }
    fb[224 + h * 4 + p] = sqrtf(s2 + EPSV);    // res_norm block [224,256)
  }
}

// ---------------- K9: output projection feat(512x1280) @ W_out --------------
__global__ void k9_out(const float* __restrict__ feat, const float* __restrict__ WoutT,
                       const float* __restrict__ bout, float* __restrict__ out) {
  const int gw = blockIdx.x * (blockDim.x >> 5) + (threadIdx.x >> 5);
  if (gw >= 32 * 16) return;
  const int m0 = (gw >> 4) * 16, n0 = (gw & 15) * 16;
  v8f acc = {};
#pragma unroll 4
  for (int k0 = 0; k0 < 1280; k0 += 32)
    acc = wmma_f16(frag_a_full(feat, 1280, m0, k0),
                   frag_bT_full(WoutT, 1280, k0, n0), acc);

  const int lane = threadIdx.x & 31;
  const int n = n0 + (lane & 15), rb = (lane >> 4) * 8;
  const float bb = bout[n];
#pragma unroll
  for (int r = 0; r < 8; ++r)
    out[(size_t)(m0 + rb + r) * NODE_D + n] = acc[r] + bb;
}

// ---------------- host launch ----------------------------------------------
extern "C" void kernel_launch(void* const* d_in, const int* in_sizes, int n_in,
                              void* d_out, int out_size, void* d_ws, size_t ws_size,
                              hipStream_t stream) {
  const float* node  = (const float*)d_in[0];
  const float* edge  = (const float*)d_in[1];
  const float* rot   = (const float*)d_in[2];
  const float* trans = (const float*)d_in[3];
  const float* Wsq   = (const float*)d_in[4];
  const float* Wsk   = (const float*)d_in[5];
  const float* Wsv   = (const float*)d_in[6];
  const float* Wpq   = (const float*)d_in[7];
  const float* Wpk   = (const float*)d_in[8];
  const float* Wpv   = (const float*)d_in[9];
  const float* pwts  = (const float*)d_in[10];
  const float* Wpb   = (const float*)d_in[11];
  const float* bpb   = (const float*)d_in[12];
  const float* Wout  = (const float*)d_in[13];
  const float* bout  = (const float*)d_in[14];
  float* out = (float*)d_out;
  float* ws  = (float*)d_ws;

  // workspace layout (float offsets, all 16-float aligned)
  float* qs    = ws;               // 8*512*32 padded            (131072)
  float* ks    = qs    + 131072;   // 8*512*32 padded            (131072)
  float* qp    = ks    + 131072;   // 8*512*32 padded            (131072)
  float* kp    = qp    + 131072;   // 8*512*32 padded            (131072)
  float* vpT   = kp    + 131072;   // [H][16][N], rows 12..15 0  (65536)
  // ---- end of zeroed region: 589824 floats
  float* vsT   = vpT   + 65536;    // [H][16][N]                 (65536)
  float* prq   = vsT   + 65536;    // 512*96                     (49152)
  float* prk   = prq   + 49152;
  float* prv   = prk   + 49152;
  float* q2    = prv   + 49152;    // 8*512                      (4096)
  float* k2    = q2    + 4096;
  float* ptw   = k2    + 4096;     // 8*512*12                   (49152)
  float* WpbT  = ptw   + 49152;    // 16*128                     (2048)
  float* WT    = WpbT  + 2048;     // 672*256                    (172032)
  float* WoutT = WT    + 172032;   // 256*1280                   (327680)
  float* feat  = WoutT + 327680;   // 512*1280                   (655360)
  float* L     = feat  + 655360;   // 8*512*512 attn             (2097152)

  const int TB = 128;  // 4 waves per block
  k0_init<<<(589824 + 255) / 256, 256, 0, stream>>>(qs, Wsq, Wsk, Wsv, Wpq, Wpk, Wpv,
                                                    WT, Wpb, WpbT, Wout, WoutT);
  k1_proj<<<(1344 + 3) / 4, TB, 0, stream>>>(node, WT, qs, ks, vsT, prq, prk, prv);
  k2_transform<<<(NRES * HEADS + 255) / 256, 256, 0, stream>>>(prq, prk, prv, rot, trans,
                                                               qp, kp, vpT, q2, k2);
  k3_pairbias<<<4096 / 4, TB, 0, stream>>>(edge, WpbT, bpb, L);
  k4_logits<<<8192 / 4, TB, 0, stream>>>(qs, ks, qp, kp, q2, k2, pwts, L);
  k5_softmax<<<4096 / 4, TB, 0, stream>>>(L);
  k6_apply_v<<<256 / 4, TB, 0, stream>>>(L, vsT, vpT, feat, ptw);
  k7_pair<<<4096 / 4, TB, 0, stream>>>(L, edge, feat);
  k8_points_out<<<(NRES * HEADS + 255) / 256, 256, 0, stream>>>(ptw, rot, trans, feat);
  k9_out<<<512 / 4, TB, 0, stream>>>(feat, WoutT, bout, out);
}
